// MultiHeadAttention_79018808312395
// MI455X (gfx1250) — matrix operand order repaired and checked by reading, not compiled, not run on hardware
//
#include <hip/hip_runtime.h>
#include <hip/hip_bf16.h>

// MHA forward for MI455X (gfx1250, wave32, WMMA f16 path + async LDS copies).
// B=2, S=2048, E=1024, H=16, D=64.

typedef _Float16 f16;
typedef __attribute__((ext_vector_type(16))) _Float16 v16h;
typedef __attribute__((ext_vector_type(8)))  _Float16 f16x8;
typedef __attribute__((ext_vector_type(4)))  _Float16 f16x4;
typedef __attribute__((ext_vector_type(8)))  float    v8f;
typedef __attribute__((ext_vector_type(4)))  float    f32x4;

#define MB 2
#define MS 2048
#define ME 1024
#define MH 16
#define MD 64

__device__ __forceinline__ v8f wmma_f16(v16h a, v16h b, v8f c) {
  return __builtin_amdgcn_wmma_f32_16x16x32_f16(
      /*neg_a=*/false, a, /*neg_b=*/false, b,
      /*c_mod=*/(short)0, c, /*reuse_a=*/false, /*reuse_b=*/false);
}

// Hardware reciprocal (v_rcp_f32, ~1 ulp) — plenty downstream of f16 WMMA.
__device__ __forceinline__ float fast_rcp(float x) {
  return __builtin_amdgcn_rcpf(x);
}

// Async global->LDS 128-bit copy per lane (CDNA5 ASYNCcnt path).
// Generic shared pointers carry the LDS byte offset in their low 32 bits.
__device__ __forceinline__ void async_ld_b128(void* lds, const void* gaddr) {
  asm volatile("global_load_async_to_lds_b128 %0, %1, off"
               :: "v"((unsigned int)(unsigned long long)lds), "v"(gaddr)
               : "memory");
}
__device__ __forceinline__ void wait_async0() {
  asm volatile("s_wait_asynccnt 0x0" ::: "memory");
}

// A-operand fragment (16x32 f16, MxK): lane L -> row L&15,
// K chunks {kb..kb+7, kb+16..kb+23}, kb=(L>>4)*8.  base points at [row0][k0].
__device__ __forceinline__ v16h load_frag_a(const f16* base, int ld, int lane) {
  const f16* p = base + (size_t)(lane & 15) * ld + ((lane >> 4) * 8);
  f16x8 lo = *(const f16x8*)(p);
  f16x8 hi = *(const f16x8*)(p + 16);
  return __builtin_shufflevector(lo, hi, 0,1,2,3,4,5,6,7,8,9,10,11,12,13,14,15);
}

// B-operand fragment (32x16 f16, KxN) sourced from memory holding B^T row-major
// ([N][K]): lane L -> column L&15, K chunks {kb..kb+7, kb+16..kb+23}, kb=(L>>4)*8.
__device__ __forceinline__ v16h load_frag_b(const f16* base, int ld, int lane) {
  const f16* p = base + (size_t)(lane & 15) * ld + ((lane >> 4) * 8);
  f16x8 lo = *(const f16x8*)(p);
  f16x8 hi = *(const f16x8*)(p + 16);
  return __builtin_shufflevector(lo, hi, 0,1,2,3,4,5,6,7,8,9,10,11,12,13,14,15);
}

// C = A @ Bw^T + bias   (A: [M][K], Bw: [N][K] both row-major; NT GEMM)
// A_F16: A is f16 (staged via async LDS copy), else f32 (converted on stage).
// OUT_F16: C stored f16, else f32.
template <int A_F16, int OUT_F16>
__global__ __launch_bounds__(128) void gemm_nt_kernel(
    const void* __restrict__ Ain, const float* __restrict__ Bw,
    const float* __restrict__ bias, void* __restrict__ Cout,
    int M, int N, int K) {
  __shared__ f16 As[64][40];
  __shared__ f16 Bs[64][40];

  const int lane = threadIdx.x & 31;
  const int w    = threadIdx.x >> 5;
  const int m0   = blockIdx.y * 64;
  const int n0   = blockIdx.x * 64;
  const int wrow = (w >> 1) * 32;
  const int wcol = (w & 1) * 32;

  v8f acc[2][2] = {};

  const int t    = threadIdx.x;
  const int lrow = t >> 1;         // 0..63
  const int lcb  = (t & 1) * 16;   // 0 or 16 (16 elems per thread per tile)

  for (int k0 = 0; k0 < K; k0 += 32) {
    __syncthreads();
    // ---- stage A tile (64 x 32) as f16 ----
    if (A_F16) {
      const f16* g = (const f16*)Ain + (size_t)(m0 + lrow) * K + k0 + lcb;
      async_ld_b128(&As[lrow][lcb + 0], g + 0);
      async_ld_b128(&As[lrow][lcb + 8], g + 8);
    } else {
      const float* g = (const float*)Ain + (size_t)(m0 + lrow) * K + k0 + lcb;
#pragma unroll
      for (int j = 0; j < 4; ++j) {
        f32x4 v = ((const f32x4*)g)[j];
        f16x4 h = {(f16)v.x, (f16)v.y, (f16)v.z, (f16)v.w};
        *(f16x4*)&As[lrow][lcb + j * 4] = h;
      }
    }
    // ---- stage B tile (64 rows of Bw x 32) as f16 ----
    {
      const float* g = Bw + (size_t)(n0 + lrow) * K + k0 + lcb;
#pragma unroll
      for (int j = 0; j < 4; ++j) {
        f32x4 v = ((const f32x4*)g)[j];
        f16x4 h = {(f16)v.x, (f16)v.y, (f16)v.z, (f16)v.w};
        *(f16x4*)&Bs[lrow][lcb + j * 4] = h;
      }
    }
    if (A_F16) wait_async0();
    __syncthreads();

    v16h af0 = load_frag_a(&As[wrow][0], 40, lane);
    v16h af1 = load_frag_a(&As[wrow + 16][0], 40, lane);
    v16h bf0 = load_frag_b(&Bs[wcol][0], 40, lane);
    v16h bf1 = load_frag_b(&Bs[wcol + 16][0], 40, lane);
    acc[0][0] = wmma_f16(af0, bf0, acc[0][0]);
    acc[0][1] = wmma_f16(af0, bf1, acc[0][1]);
    acc[1][0] = wmma_f16(af1, bf0, acc[1][0]);
    acc[1][1] = wmma_f16(af1, bf1, acc[1][1]);
  }

  // ---- epilogue: bias add + store ----
#pragma unroll
  for (int tc = 0; tc < 2; ++tc) {
    const int col = n0 + wcol + tc * 16 + (lane & 15);
    const float bv = bias[col];
#pragma unroll
    for (int tr = 0; tr < 2; ++tr) {
#pragma unroll
      for (int r = 0; r < 8; ++r) {
        const int row = m0 + wrow + tr * 16 + ((lane >> 4) * 8) + r;
        const float val = acc[tr][tc][r] + bv;
        if (OUT_F16)
          ((f16*)Cout)[(size_t)row * N + col] = (f16)val;
        else
          ((float*)Cout)[(size_t)row * N + col] = val;
      }
    }
  }
}

// Streaming sigmoid-softmax attention, one block per (b, h, 64-query tile).
// Since sigmoid(w) in (0,1), softmax needs no running-max: p = exp(w) in (1,e).
__global__ __launch_bounds__(128) void attn_kernel(
    const f16* __restrict__ Q, const f16* __restrict__ Kc,
    const f16* __restrict__ Vc, f16* __restrict__ Oc,
    const int* __restrict__ indp) {
  const int lane = threadIdx.x & 31;
  const int w    = threadIdx.x >> 5;
  const int qt   = blockIdx.x % (MS / 64);
  const int h    = (blockIdx.x / (MS / 64)) % MH;
  const int b    = blockIdx.x / ((MS / 64) * MH);
  const int qbase = qt * 64;
  const int ind   = *indp;

  __shared__ f16 Qs[64][72];       // [q][d]
  __shared__ f16 Ks[64][72];       // [key][d]
  __shared__ f16 Vt[64][72];       // [d][key]  (transposed for PV B-operand)
  __shared__ f16 Ps[4][16][72];    // per-wave exp(w) strip [q][key]

  const int t   = threadIdx.x;
  const int row = t >> 1;          // 0..63
  const int cb  = (t & 1) * 32;    // 0 or 32

  // ---- stage Q tile (64 q-rows x 64 d) via async LDS copy ----
  {
    const f16* g = Q + ((size_t)(b * MS + qbase + row)) * ME + h * MD + cb;
#pragma unroll
    for (int j = 0; j < 4; ++j) async_ld_b128(&Qs[row][cb + j * 8], g + j * 8);
  }
  wait_async0();
  __syncthreads();

  // Preload this wave's Q A-fragments (rows w*16..+15), K chunks d=0..31, 32..63
  const v16h aq0 = load_frag_a(&Qs[w * 16][0], 72, lane);
  const v16h aq1 = load_frag_a(&Qs[w * 16][32], 72, lane);

  v8f  oacc[4] = {};
  float den[8] = {0.f, 0.f, 0.f, 0.f, 0.f, 0.f, 0.f, 0.f};

  for (int kt = 0; kt < MS / 64; ++kt) {
    __syncthreads();  // protect Ks/Vt against previous iteration's readers
    // ---- stage K tile (async) and V tile transposed (manual) ----
    {
      const size_t gro = ((size_t)(b * MS + kt * 64 + row)) * ME + h * MD + cb;
      const f16* gk = Kc + gro;
      const f16* gv = Vc + gro;
#pragma unroll
      for (int j = 0; j < 4; ++j) async_ld_b128(&Ks[row][cb + j * 8], gk + j * 8);
#pragma unroll
      for (int j = 0; j < 4; ++j) {
        f16x8 vv = ((const f16x8*)gv)[j];
#pragma unroll
        for (int i = 0; i < 8; ++i) Vt[cb + j * 8 + i][row] = vv[i];
      }
      // Prefetch next key tile into cache (global_prefetch_b8).
      if (kt + 1 < MS / 64) {
        const f16* pk = gk + (size_t)64 * ME;
        const f16* pv = gv + (size_t)64 * ME;
        __builtin_prefetch(pk, 0, 1);
        __builtin_prefetch(pv, 0, 1);
      }
    }
    wait_async0();
    __syncthreads();

    // ---- scores: 16 q-rows x 64 keys per wave, then sigmoid+exp ----
    v8f sacc[4];
#pragma unroll
    for (int nt = 0; nt < 4; ++nt) {
      v8f acc = {};
      v16h b0 = load_frag_b(&Ks[nt * 16][0], 72, lane);
      acc = wmma_f16(aq0, b0, acc);
      v16h b1 = load_frag_b(&Ks[nt * 16][32], 72, lane);
      acc = wmma_f16(aq1, b1, acc);
#pragma unroll
      for (int r = 0; r < 8; ++r) {
        float s  = acc[r] * 0.125f;                   // 1/sqrt(D), D=64
        float sg = fast_rcp(1.f + __expf(-s));        // sigmoid via v_rcp_f32
        if (ind) sg = 1.f - sg;
        float p = __expf(sg);                         // softmax numerator, (1,e)
        acc[r] = p;
        Ps[w][(lane >> 4) * 8 + r][nt * 16 + (lane & 15)] = (f16)p;
      }
      sacc[nt] = acc;
    }

    // ---- denominator: butterfly row-sum across the 16 lanes of each half ----
#pragma unroll
    for (int r = 0; r < 8; ++r) {
      float part = sacc[0][r] + sacc[1][r] + sacc[2][r] + sacc[3][r];
      part += __shfl_xor(part, 1, 32);
      part += __shfl_xor(part, 2, 32);
      part += __shfl_xor(part, 4, 32);
      part += __shfl_xor(part, 8, 32);
      den[r] += part;
    }

    // ---- PV: oacc += P(16x64) @ V(64x64); P-fragment hoisted over dt ----
#pragma unroll
    for (int ks2 = 0; ks2 < 64; ks2 += 32) {
      v16h pa = load_frag_a(&Ps[w][0][ks2], 72, lane);
#pragma unroll
      for (int dt = 0; dt < 4; ++dt) {
        v16h vb = load_frag_b(&Vt[dt * 16][ks2], 72, lane);
        oacc[dt] = wmma_f16(pa, vb, oacc[dt]);
      }
    }
  }

  // ---- normalize (one v_rcp per row) + store f16 [B,S,E] ----
#pragma unroll
  for (int r = 0; r < 8; ++r) {
    const float inv = fast_rcp(den[r]);
    const int qrow = qbase + w * 16 + ((lane >> 4) * 8) + r;
    const size_t rbase = ((size_t)(b * MS + qrow)) * ME + h * MD + (lane & 15);
#pragma unroll
    for (int dt = 0; dt < 4; ++dt) {
      Oc[rbase + dt * 16] = (f16)(oacc[dt][r] * inv);
    }
  }
}

extern "C" void kernel_launch(void* const* d_in, const int* in_sizes, int n_in,
                              void* d_out, int out_size, void* d_ws, size_t ws_size,
                              hipStream_t stream) {
  (void)in_sizes; (void)n_in; (void)out_size; (void)ws_size;
  const float* queries = (const float*)d_in[0];
  const float* keys    = (const float*)d_in[1];
  const float* values  = (const float*)d_in[2];
  const float* Wq = (const float*)d_in[3];
  const float* bq = (const float*)d_in[4];
  const float* Wk = (const float*)d_in[5];
  const float* bk = (const float*)d_in[6];
  const float* Wv = (const float*)d_in[7];
  const float* bv = (const float*)d_in[8];
  const float* Wo = (const float*)d_in[9];
  const float* bo = (const float*)d_in[10];
  const int*   ind = (const int*)d_in[11];

  const size_t elems = (size_t)MB * MS * ME;  // 4 Mi elems
  f16* Q16 = (f16*)d_ws;
  f16* K16 = Q16 + elems;
  f16* V16 = K16 + elems;
  f16* A16 = V16 + elems;   // total 32 MB f16 scratch

  const int M = MB * MS;    // 4096
  dim3 grid(ME / 64, M / 64);  // (16, 64)
  dim3 blk(128);

  gemm_nt_kernel<0, 1><<<grid, blk, 0, stream>>>(queries, Wq, bq, Q16, M, ME, ME);
  gemm_nt_kernel<0, 1><<<grid, blk, 0, stream>>>(keys,    Wk, bk, K16, M, ME, ME);
  gemm_nt_kernel<0, 1><<<grid, blk, 0, stream>>>(values,  Wv, bv, V16, M, ME, ME);

  dim3 agrid(MB * MH * (MS / 64));  // 1024 blocks
  attn_kernel<<<agrid, blk, 0, stream>>>(Q16, K16, V16, A16, ind);

  gemm_nt_kernel<1, 0><<<grid, blk, 0, stream>>>(A16, Wo, bo, (float*)d_out, M, ME, ME);
}
